// BaselineModel_71055938945179
// MI455X (gfx1250) — matrix operand order repaired and checked by reading, not compiled, not run on hardware
//
#include <hip/hip_runtime.h>
#include <hip/hip_bf16.h>

// ---------------------------------------------------------------------------
// Model constants (from reference)
// ---------------------------------------------------------------------------
#define BB      256
#define LL      200
#define HH      128
#define NHH     2
#define DHH     64
#define NBB     2
#define DFFF    512
#define H4V     32
#define F_ITEMC 6
#define N_CBC   4
#define V_FEAT1 100001   // V_FEAT + 1
#define CB_SZ1  8193     // CB_SIZE + 1

typedef __attribute__((ext_vector_type(16))) __bf16 v16bf;
typedef __attribute__((ext_vector_type(8)))  __bf16 bf16x8;
typedef __attribute__((ext_vector_type(8)))  float  v8f;

#if defined(__has_builtin)
#if __has_builtin(__builtin_amdgcn_sched_barrier)
#define SCHED_FENCE() __builtin_amdgcn_sched_barrier(0)
#endif
#endif
#ifndef SCHED_FENCE
#define SCHED_FENCE()
#endif

// ---------------------------------------------------------------------------
// Embedding: x[b,l,:] = item_emb[seq] + pos_table[l+1] + tile4(sparse sums)
// also writes key-padding mask (-1e9 where seq==0).  One wave per token.
// ---------------------------------------------------------------------------
__global__ __launch_bounds__(256) void embed_kernel(
    const int* __restrict__ seq, const int* __restrict__ seq_feat,
    const int* __restrict__ seq_sem, const float* __restrict__ item_emb,
    const float* __restrict__ pos_table, const float* __restrict__ ist,
    const float* __restrict__ sem, float* __restrict__ X,
    float* __restrict__ maskb)
{
    int t    = blockIdx.x * 8 + (threadIdx.x >> 5);   // b*LL + l
    int lane = threadIdx.x & 31;
    if (t >= BB * LL) return;
    int l  = t % LL;
    int id = seq[t];
    float sp = 0.f;
    #pragma unroll
    for (int f = 0; f < F_ITEMC; ++f) {
        int idx = seq_feat[t * F_ITEMC + f];
        sp += ist[((size_t)f * V_FEAT1 + idx) * H4V + lane];
    }
    #pragma unroll
    for (int c = 0; c < N_CBC; ++c) {
        int idx = seq_sem[t * N_CBC + c];
        sp += sem[((size_t)c * CB_SZ1 + idx) * H4V + lane];
    }
    const float* ie = item_emb + (size_t)id * HH;
    const float* pt = pos_table + (size_t)(l + 1) * HH;
    #pragma unroll
    for (int j = 0; j < 4; ++j) {
        int h = lane + 32 * j;                 // h % 32 == lane  (tile4)
        X[(size_t)t * HH + h] = ie[h] + pt[h] + sp;
    }
    if (lane == 0) maskb[t] = (id == 0) ? -1e9f : 0.f;
}

// ---------------------------------------------------------------------------
// Weight prep: W[K,N] f32 -> WT[N,K] bf16 (transposed for contiguous B-frags)
// ---------------------------------------------------------------------------
__global__ __launch_bounds__(256) void wcvt_kernel(
    const float* __restrict__ W, __bf16* __restrict__ WT, int K, int N)
{
    int idx = blockIdx.x * 256 + threadIdx.x;
    if (idx >= K * N) return;
    int k = idx / N, n = idx % N;
    WT[(size_t)n * K + k] = (__bf16)W[idx];
}

// ---------------------------------------------------------------------------
// LayerNorm: one wave per row of 128, bf16 output (GEMM A operand)
// ---------------------------------------------------------------------------
__global__ __launch_bounds__(256) void ln_kernel(
    const float* __restrict__ X, const float* __restrict__ g,
    const float* __restrict__ bt, __bf16* __restrict__ Y)
{
    int row  = blockIdx.x * 8 + (threadIdx.x >> 5);
    int lane = threadIdx.x & 31;
    if (row >= BB * LL) return;
    const float* x = X + (size_t)row * HH;
    float v[4];
    float s = 0.f;
    #pragma unroll
    for (int j = 0; j < 4; ++j) { v[j] = x[lane + 32 * j]; s += v[j]; }
    #pragma unroll
    for (int off = 16; off >= 1; off >>= 1) s += __shfl_xor(s, off, 32);
    float mean = s * (1.f / HH);
    float vs = 0.f;
    #pragma unroll
    for (int j = 0; j < 4; ++j) { float d = v[j] - mean; vs += d * d; }
    #pragma unroll
    for (int off = 16; off >= 1; off >>= 1) vs += __shfl_xor(vs, off, 32);
    float inv = rsqrtf(vs * (1.f / HH) + 1e-5f);
    #pragma unroll
    for (int j = 0; j < 4; ++j) {
        int h = lane + 32 * j;
        Y[(size_t)row * HH + h] = (__bf16)((v[j] - mean) * inv * g[h] + bt[h]);
    }
}

// ---------------------------------------------------------------------------
// WMMA GEMM: Y[M,N] = A[M,K](bf16) @ WT[N,K](bf16)^T + bias (+res) (ReLU opt)
// One wave computes a 16x64 strip (4 accumulators).  Fragments are double-
// buffered and the prefetch / WMMA phases are separated by sched_barrier(0)
// so the machine scheduler cannot sink the k+1 loads below the k WMMAs:
// the matrix pipe overlaps with L2 fetch inside a single wave.
// All fragment loads are contiguous b128s in the native ISA 7.12.2 layouts.
// EXEC all-ones in every WMMA wave.
// ---------------------------------------------------------------------------
template <int K, int N, bool RELU, bool RES, typename OutT>
__global__ __launch_bounds__(256) void gemm_wmma(
    const __bf16* __restrict__ A, const __bf16* __restrict__ WT,
    const float* __restrict__ bias, const float* __restrict__ resid,
    OutT* __restrict__ Y, int M)
{
    constexpr int NT = 4;                       // 4 n-tiles of 16 per wave
    constexpr int NK = K / 32;                  // k-steps
    int wave = threadIdx.x >> 5;
    int lane = threadIdx.x & 31;
    int m0 = (blockIdx.x * 8 + wave) * 16;
    int n0 = blockIdx.y * (16 * NT);
    if (m0 >= M) return;                        // wave-uniform
    int half = lane >> 4;                       // 0 or 1
    int l16  = lane & 15;
    const __bf16* arow = A + (size_t)(m0 + l16) * K + half * 8;
    const __bf16* brow[NT];
    #pragma unroll
    for (int t = 0; t < NT; ++t)
        brow[t] = WT + (size_t)(n0 + t * 16 + l16) * K + half * 8;

    auto loadA = [&](int k0, v16bf& dst) {
        bf16x8 a0 = *(const bf16x8*)(arow + k0);
        bf16x8 a1 = *(const bf16x8*)(arow + k0 + 16);
        #pragma unroll
        for (int j = 0; j < 8; ++j) { dst[j] = a0[j]; dst[8 + j] = a1[j]; }
    };
    auto loadB = [&](int k0, int t, v16bf& dst) {
        bf16x8 b0 = *(const bf16x8*)(brow[t] + k0);
        bf16x8 b1 = *(const bf16x8*)(brow[t] + k0 + 16);
        #pragma unroll
        for (int j = 0; j < 8; ++j) { dst[j] = b0[j]; dst[8 + j] = b1[j]; }
    };

    v16bf afr[2];
    v16bf bfr[2][NT];
    loadA(0, afr[0]);
    #pragma unroll
    for (int t = 0; t < NT; ++t) loadB(0, t, bfr[0][t]);

    v8f zero = {};
    v8f acc[NT];
    #pragma unroll
    for (int t = 0; t < NT; ++t) acc[t] = zero;

    #pragma unroll
    for (int i = 0; i < NK; ++i) {
        const int cur = i & 1, nxt = cur ^ 1;
        if (i + 1 < NK) {
            loadA((i + 1) * 32, afr[nxt]);
            #pragma unroll
            for (int t = 0; t < NT; ++t) loadB((i + 1) * 32, t, bfr[nxt][t]);
        }
        SCHED_FENCE();                          // keep prefetch above WMMAs
        #pragma unroll
        for (int t = 0; t < NT; ++t)
            acc[t] = __builtin_amdgcn_wmma_f32_16x16x32_bf16(
                false, afr[cur], false, bfr[cur][t], (short)0, acc[t],
                false, false);
        SCHED_FENCE();                          // keep WMMAs above next loads
    }

    // Epilogue: C/D layout -> lane, VGPR j holds (m0 + j + 8*half, n)
    #pragma unroll
    for (int t = 0; t < NT; ++t) {
        int n = n0 + t * 16 + l16;
        float bv = bias[n];
        #pragma unroll
        for (int j = 0; j < 8; ++j) {
            int m = m0 + j + 8 * half;
            float val = acc[t][j] + bv;
            if (RES) val += resid[(size_t)m * N + n];
            if (RELU) val = fmaxf(val, 0.f);
            Y[(size_t)m * N + n] = (OutT)val;
        }
    }
}

// ---------------------------------------------------------------------------
// Attention: one workgroup per (b, head).  Q/K/V arrive as bf16; K/V staged
// in LDS (50 KB).  One thread per query row, single-pass online softmax in
// f32, bf16 output (consumed only by the o-projection GEMM).
// ---------------------------------------------------------------------------
__global__ __launch_bounds__(256) void attn_kernel(
    const __bf16* __restrict__ Q, const __bf16* __restrict__ Kb,
    const __bf16* __restrict__ Vb, const float* __restrict__ maskb,
    __bf16* __restrict__ O)
{
    __shared__ __bf16 ks[LL][DHH];
    __shared__ __bf16 vs[LL][DHH];
    __shared__ float  mk[LL];
    int bh = blockIdx.x;
    int b = bh >> 1, h = bh & 1;
    const size_t base = (size_t)b * LL * HH + (size_t)h * DHH;
    for (int i = threadIdx.x; i < LL * DHH; i += 256) {
        int r = i / DHH, c = i % DHH;
        ks[r][c] = Kb[base + (size_t)r * HH + c];
        vs[r][c] = Vb[base + (size_t)r * HH + c];
    }
    for (int i = threadIdx.x; i < LL; i += 256) mk[i] = maskb[b * LL + i];
    __syncthreads();
    int qr = threadIdx.x;
    if (qr < LL) {
        float qreg[DHH];
        #pragma unroll
        for (int d = 0; d < DHH; ++d)
            qreg[d] = (float)Q[base + (size_t)qr * HH + d] * 0.125f; // 1/sqrt(64)
        float o[DHH] = {};
        float mmax = -1e30f, ssum = 0.f;
        for (int k = 0; k < LL; ++k) {
            float dot = 0.f;
            #pragma unroll
            for (int d = 0; d < DHH; ++d) dot += qreg[d] * (float)ks[k][d];
            dot += mk[k];
            float nm   = fmaxf(mmax, dot);
            float corr = __expf(mmax - nm);
            float w    = __expf(dot - nm);
            ssum = ssum * corr + w;
            #pragma unroll
            for (int d = 0; d < DHH; ++d) o[d] = o[d] * corr + w * (float)vs[k][d];
            mmax = nm;
        }
        float inv = 1.f / ssum;
        #pragma unroll
        for (int d = 0; d < DHH; ++d)
            O[base + (size_t)qr * HH + d] = (__bf16)(o[d] * inv);
    }
}

// ---------------------------------------------------------------------------
// Final: last-token extraction + pos/neg embedding dot products (f32 x)
// ---------------------------------------------------------------------------
__global__ __launch_bounds__(32) void final_kernel(
    const int* __restrict__ seq, const int* __restrict__ pos,
    const int* __restrict__ neg, const int* __restrict__ pos_feat,
    const int* __restrict__ neg_feat, const int* __restrict__ pos_sem,
    const int* __restrict__ neg_sem, const float* __restrict__ item_emb,
    const float* __restrict__ ist, const float* __restrict__ sem,
    const float* __restrict__ X, float* __restrict__ out)
{
    int b = blockIdx.x;
    int lane = threadIdx.x;
    int cnt = 0;
    for (int l = lane; l < LL; l += 32) cnt += (seq[b * LL + l] != 0) ? 1 : 0;
    #pragma unroll
    for (int off = 16; off >= 1; off >>= 1) cnt += __shfl_xor(cnt, off, 32);
    int last = cnt - 1;
    const float* xl = X + ((size_t)b * LL + last) * HH;
    float spp = 0.f, spn = 0.f;
    #pragma unroll
    for (int f = 0; f < F_ITEMC; ++f) {
        spp += ist[((size_t)f * V_FEAT1 + pos_feat[b * F_ITEMC + f]) * H4V + lane];
        spn += ist[((size_t)f * V_FEAT1 + neg_feat[b * F_ITEMC + f]) * H4V + lane];
    }
    #pragma unroll
    for (int c = 0; c < N_CBC; ++c) {
        spp += sem[((size_t)c * CB_SZ1 + pos_sem[b * N_CBC + c]) * H4V + lane];
        spn += sem[((size_t)c * CB_SZ1 + neg_sem[b * N_CBC + c]) * H4V + lane];
    }
    int pid = pos[b * LL + LL - 1];
    int nid = neg[b * LL + LL - 1];
    const float* pe = item_emb + (size_t)pid * HH;
    const float* ne = item_emb + (size_t)nid * HH;
    float dp = 0.f, dn = 0.f;
    #pragma unroll
    for (int j = 0; j < 4; ++j) {
        int h = lane + 32 * j;
        float xv = xl[h];
        dp += xv * (pe[h] + spp);
        dn += xv * (ne[h] + spn);
    }
    #pragma unroll
    for (int off = 16; off >= 1; off >>= 1) {
        dp += __shfl_xor(dp, off, 32);
        dn += __shfl_xor(dn, off, 32);
    }
    if (lane == 0) { out[b] = dp; out[BB + b] = dn; }
}

// ---------------------------------------------------------------------------
// Host launcher
// ---------------------------------------------------------------------------
extern "C" void kernel_launch(void* const* d_in, const int* in_sizes, int n_in,
                              void* d_out, int out_size, void* d_ws, size_t ws_size,
                              hipStream_t stream) {
    (void)in_sizes; (void)n_in; (void)out_size; (void)ws_size;
    const int*   seq       = (const int*)  d_in[0];
    const int*   pos       = (const int*)  d_in[1];
    const int*   neg       = (const int*)  d_in[2];
    const int*   seq_feat  = (const int*)  d_in[6];
    const int*   pos_feat  = (const int*)  d_in[7];
    const int*   neg_feat  = (const int*)  d_in[8];
    const int*   seq_sem   = (const int*)  d_in[9];
    const int*   pos_sem   = (const int*)  d_in[10];
    const int*   neg_sem   = (const int*)  d_in[11];
    const float* item_emb  = (const float*)d_in[12];
    const float* pos_table = (const float*)d_in[13];
    const float* ist       = (const float*)d_in[14];
    const float* sem       = (const float*)d_in[15];
    const float* wq = (const float*)d_in[16]; const float* bq = (const float*)d_in[17];
    const float* wk = (const float*)d_in[18]; const float* bk = (const float*)d_in[19];
    const float* wv = (const float*)d_in[20]; const float* bv = (const float*)d_in[21];
    const float* wo = (const float*)d_in[22]; const float* bo = (const float*)d_in[23];
    const float* ln1_g = (const float*)d_in[24]; const float* ln1_b = (const float*)d_in[25];
    const float* ln2_g = (const float*)d_in[26]; const float* ln2_b = (const float*)d_in[27];
    const float* w1 = (const float*)d_in[28]; const float* b1 = (const float*)d_in[29];
    const float* w2 = (const float*)d_in[30]; const float* b2 = (const float*)d_in[31];
    float* out = (float*)d_out;

    const int    M  = BB * LL;               // 51200
    const size_t SZ = (size_t)M * HH;        // 6,553,600 elements

    // ---- workspace carve-up (byte offsets kept 256B aligned) ----
    char* base = (char*)d_ws;
    size_t off = 0;
    auto carve = [&](size_t bytes) {
        char* p = base + off;
        off += (bytes + 255) & ~(size_t)255;
        return p;
    };
    float*  x    = (float*) carve(SZ * 4);
    __bf16* y    = (__bf16*)carve(SZ * 2);
    __bf16* qb   = (__bf16*)carve(SZ * 2);
    __bf16* kb   = (__bf16*)carve(SZ * 2);
    __bf16* vb   = (__bf16*)carve(SZ * 2);
    __bf16* t0   = (__bf16*)carve(SZ * 2);
    __bf16* hbuf = (__bf16*)carve((size_t)M * DFFF * 2);
    float*  mk   = (float*) carve((size_t)M * 4);
    // bf16 transposed weights, per block
    __bf16* wT[NBB][6];
    for (int l = 0; l < NBB; ++l) {
        wT[l][0] = (__bf16*)carve((size_t)HH * HH * 2);    // wq^T
        wT[l][1] = (__bf16*)carve((size_t)HH * HH * 2);    // wk^T
        wT[l][2] = (__bf16*)carve((size_t)HH * HH * 2);    // wv^T
        wT[l][3] = (__bf16*)carve((size_t)HH * HH * 2);    // wo^T
        wT[l][4] = (__bf16*)carve((size_t)HH * DFFF * 2);  // w1^T  [512,128]
        wT[l][5] = (__bf16*)carve((size_t)DFFF * HH * 2);  // w2^T  [128,512]
    }

    const dim3 blk256(256);
    const int  rowBlocks = (M + 7) / 8;              // wave-per-row kernels
    const dim3 gN128(M / 128, HH / 64);              // 400 x 2  (N=128, 64 cols/wave)
    const dim3 gN512(M / 128, DFFF / 64);            // 400 x 8  (N=512)
    const int  wgrid = (HH * HH + 255) / 256;        // 64
    const int  wgridF = (HH * DFFF + 255) / 256;     // 256

    // ---- weight prep (bf16 transpose) ----
    for (int l = 0; l < NBB; ++l) {
        wcvt_kernel<<<wgrid,  blk256, 0, stream>>>(wq + (size_t)l * HH * HH,   wT[l][0], HH, HH);
        wcvt_kernel<<<wgrid,  blk256, 0, stream>>>(wk + (size_t)l * HH * HH,   wT[l][1], HH, HH);
        wcvt_kernel<<<wgrid,  blk256, 0, stream>>>(wv + (size_t)l * HH * HH,   wT[l][2], HH, HH);
        wcvt_kernel<<<wgrid,  blk256, 0, stream>>>(wo + (size_t)l * HH * HH,   wT[l][3], HH, HH);
        wcvt_kernel<<<wgridF, blk256, 0, stream>>>(w1 + (size_t)l * HH * DFFF, wT[l][4], HH, DFFF);
        wcvt_kernel<<<wgridF, blk256, 0, stream>>>(w2 + (size_t)l * DFFF * HH, wT[l][5], DFFF, HH);
    }

    embed_kernel<<<rowBlocks, blk256, 0, stream>>>(
        seq, seq_feat, seq_sem, item_emb, pos_table, ist, sem, x, mk);

    for (int l = 0; l < NBB; ++l) {
        const float* bql = bq + l * HH;
        const float* bkl = bk + l * HH;
        const float* bvl = bv + l * HH;
        const float* bol = bo + l * HH;
        const float* b1l = b1 + l * DFFF;
        const float* b2l = b2 + l * HH;

        ln_kernel<<<rowBlocks, blk256, 0, stream>>>(x, ln1_g + l * HH, ln1_b + l * HH, y);
        gemm_wmma<HH, HH, false, false, __bf16><<<gN128, blk256, 0, stream>>>(
            y, wT[l][0], bql, nullptr, qb, M);
        gemm_wmma<HH, HH, false, false, __bf16><<<gN128, blk256, 0, stream>>>(
            y, wT[l][1], bkl, nullptr, kb, M);
        gemm_wmma<HH, HH, false, false, __bf16><<<gN128, blk256, 0, stream>>>(
            y, wT[l][2], bvl, nullptr, vb, M);
        attn_kernel<<<BB * NHH, blk256, 0, stream>>>(qb, kb, vb, mk, t0);
        gemm_wmma<HH, HH, false, true, float><<<gN128, blk256, 0, stream>>>(
            t0, wT[l][3], bol, x, x, M);
        ln_kernel<<<rowBlocks, blk256, 0, stream>>>(x, ln2_g + l * HH, ln2_b + l * HH, y);
        gemm_wmma<HH, DFFF, true, false, __bf16><<<gN512, blk256, 0, stream>>>(
            y, wT[l][4], b1l, nullptr, hbuf, M);
        gemm_wmma<DFFF, HH, false, true, float><<<gN128, blk256, 0, stream>>>(
            hbuf, wT[l][5], b2l, x, x, M);
    }

    final_kernel<<<BB, 32, 0, stream>>>(
        seq, pos, neg, pos_feat, neg_feat, pos_sem, neg_sem,
        item_emb, ist, sem, x, out);
}
